// MultiheadSelfAttention_33784212750425
// MI455X (gfx1250) — compile-verified, matrix operand order checked by reading, not run on hardware
//
#include <hip/hip_runtime.h>

// ---------------------------------------------------------------------------
// MI455X (gfx1250) attention block: bf16 WMMA everywhere, f32 accumulation.
// ---------------------------------------------------------------------------

typedef __attribute__((ext_vector_type(16))) __bf16 v16bf;
typedef __attribute__((ext_vector_type(8)))  float  v8f;

#define DIM     2048
#define NHEADS  16
#define HDIM    128
#define SLEN    2048
#define BATCH   2
#define ROWS    (BATCH * SLEN)          // 4096
#define QKVN    (3 * DIM)               // 6144

union AFrag { uint4 q[2]; v16bf v; };   // 16x32 bf16 A/B fragment (8 VGPRs)

__device__ __forceinline__ v8f wmma_bf16(v16bf a, v16bf b, v8f c) {
    return __builtin_amdgcn_wmma_f32_16x16x32_bf16(
        /*neg_a=*/false, a, /*neg_b=*/false, b,
        /*c_mod=*/(short)0, c, /*reuse_a=*/false, /*reuse_b=*/false);
}

__device__ __forceinline__ unsigned short f2bf(float f) {
    unsigned int u = __float_as_uint(f);
    u = (u + 0x7FFFu + ((u >> 16) & 1u)) >> 16;   // round-to-nearest-even
    return (unsigned short)u;
}

// Load an NT fragment (row-major, K contiguous): two 16B chunks at kb, kb+16.
__device__ __forceinline__ void load_frag(const unsigned short* base, size_t row,
                                          int ld, int k0, int lane, AFrag& f) {
    const unsigned short* p =
        base + (row + (size_t)(lane & 15)) * (size_t)ld + k0 + ((lane >> 4) << 3);
    const uint4* p4 = reinterpret_cast<const uint4*>(p);
    f.q[0] = p4[0];     // K = kb .. kb+7
    f.q[1] = p4[2];     // K = kb+16 .. kb+23
}

// ---------------------------------------------------------------------------
// 1) f32 -> bf16 conversion (vectorized x4)
// ---------------------------------------------------------------------------
__global__ void cvt_bf16_kernel(const float* __restrict__ src,
                                unsigned short* __restrict__ dst, int n4) {
    int i = blockIdx.x * blockDim.x + threadIdx.x;
    if (i >= n4) return;
    float4 v = reinterpret_cast<const float4*>(src)[i];
    ushort4 o;
    o.x = f2bf(v.x); o.y = f2bf(v.y); o.z = f2bf(v.z); o.w = f2bf(v.w);
    reinterpret_cast<ushort4*>(dst)[i] = o;
}

// ---------------------------------------------------------------------------
// 2) bf16 NT GEMM: C[M,N] f32 = A[M,K] * B[N,K]^T.
//    128 threads = 4 waves (2x2); each wave owns a 64x64 tile (4x4 fragments).
// ---------------------------------------------------------------------------
__global__ __launch_bounds__(128) void gemm_bf16_nt_kernel(
    const unsigned short* __restrict__ A, const unsigned short* __restrict__ B,
    float* __restrict__ C, int M, int N, int K) {
    int lane = threadIdx.x & 31;
    int w    = threadIdx.x >> 5;
    int tm   = blockIdx.y * 128 + (w >> 1) * 64;
    int tn   = blockIdx.x * 128 + (w & 1) * 64;
    int lm   = lane & 15, lh = lane >> 4;

    v8f c[4][4] = {};
    for (int k0 = 0; k0 < K; k0 += 32) {
        AFrag af[4], bf[4];
#pragma unroll
        for (int i = 0; i < 4; ++i) {
            load_frag(A, (size_t)(tm + i * 16), K, k0, lane, af[i]);
            load_frag(B, (size_t)(tn + i * 16), K, k0, lane, bf[i]);
        }
#pragma unroll
        for (int i = 0; i < 4; ++i)
#pragma unroll
            for (int j = 0; j < 4; ++j)
                c[i][j] = wmma_bf16(af[i].v, bf[j].v, c[i][j]);
    }
#pragma unroll
    for (int i = 0; i < 4; ++i)
#pragma unroll
        for (int j = 0; j < 4; ++j)
#pragma unroll
            for (int r = 0; r < 8; ++r) {
                size_t m = (size_t)(tm + i * 16 + r + 8 * lh);
                C[m * (size_t)N + tn + j * 16 + lm] = c[i][j][r];
            }
}

// ---------------------------------------------------------------------------
// 3) RMSNorm + RoPE per (b,h,s); one wave per 128-wide head row.
//    Writes q (pre-scaled by 1/sqrt(HD)), k row-major bf16; v transposed bf16.
// ---------------------------------------------------------------------------
__global__ __launch_bounds__(32) void rmsrope_kernel(
    const float* __restrict__ qkv, const float* __restrict__ cosb,
    const float* __restrict__ sinb, unsigned short* __restrict__ Qn,
    unsigned short* __restrict__ Kn, unsigned short* __restrict__ Vt) {
    int lane = threadIdx.x;
    int idx  = blockIdx.x;          // bh * SLEN + s
    int s    = idx & (SLEN - 1);
    int bh   = idx >> 11;
    int h    = bh & (NHEADS - 1);
    int b    = bh >> 4;
    int e    = lane * 4;

    const float* row = qkv + ((size_t)(b * SLEN + s)) * QKVN;
    float4 qv = *reinterpret_cast<const float4*>(row + h * HDIM + e);
    float4 kv = *reinterpret_cast<const float4*>(row + DIM + h * HDIM + e);
    float4 vv = *reinterpret_cast<const float4*>(row + 2 * DIM + h * HDIM + e);

    float sq = qv.x * qv.x + qv.y * qv.y + qv.z * qv.z + qv.w * qv.w;
    float sk = kv.x * kv.x + kv.y * kv.y + kv.z * kv.z + kv.w * kv.w;
#pragma unroll
    for (int m = 1; m < 32; m <<= 1) {
        sq += __shfl_xor(sq, m);
        sk += __shfl_xor(sk, m);
    }
    float rq = rsqrtf(sq * (1.0f / HDIM) + 1e-6f);
    float rk = rsqrtf(sk * (1.0f / HDIM) + 1e-6f);

    float c0 = cosb[s * HDIM + e],     s0 = sinb[s * HDIM + e];
    float c1 = cosb[s * HDIM + e + 2], s1 = sinb[s * HDIM + e + 2];

    const float att_scale = 0.08838834764831845f;  // 1/sqrt(128)
    size_t obase = ((size_t)bh * SLEN + s) * HDIM + e;

    {   // q: rmsnorm -> rope -> *att_scale
        float x1 = qv.x * rq, x2 = qv.y * rq, x3 = qv.z * rq, x4 = qv.w * rq;
        ushort4 o;
        o.x = f2bf((x1 * c0 - x2 * s0) * att_scale);
        o.y = f2bf((x1 * s0 + x2 * c0) * att_scale);
        o.z = f2bf((x3 * c1 - x4 * s1) * att_scale);
        o.w = f2bf((x3 * s1 + x4 * c1) * att_scale);
        *reinterpret_cast<ushort4*>(Qn + obase) = o;
    }
    {   // k: rmsnorm -> rope
        float x1 = kv.x * rk, x2 = kv.y * rk, x3 = kv.z * rk, x4 = kv.w * rk;
        ushort4 o;
        o.x = f2bf(x1 * c0 - x2 * s0);
        o.y = f2bf(x1 * s0 + x2 * c0);
        o.z = f2bf(x3 * c1 - x4 * s1);
        o.w = f2bf(x3 * s1 + x4 * c1);
        *reinterpret_cast<ushort4*>(Kn + obase) = o;
    }
    {   // v: transpose to [bh][hd][s]
        size_t vb = ((size_t)bh * HDIM + e) * SLEN + s;
        Vt[vb]            = f2bf(vv.x);
        Vt[vb + SLEN]     = f2bf(vv.y);
        Vt[vb + 2 * SLEN] = f2bf(vv.z);
        Vt[vb + 3 * SLEN] = f2bf(vv.w);
    }
}

// ---------------------------------------------------------------------------
// 4) Flash attention: one wave handles 32 query rows of one (b,h).
//    QK^T and P*V via v_wmma_f32_16x16x32_bf16; online softmax via shuffles.
// ---------------------------------------------------------------------------
__global__ __launch_bounds__(32) void attn_kernel(
    const unsigned short* __restrict__ Qn, const unsigned short* __restrict__ Kn,
    const unsigned short* __restrict__ Vt, unsigned short* __restrict__ Obf) {
    __shared__ unsigned short Pl[32 * 32];   // P tile staging (C-layout -> A-layout)

    int lane = threadIdx.x;
    int qb   = blockIdx.x & 63;              // 64 query blocks of 32
    int bh   = blockIdx.x >> 6;              // 0..31
    int b    = bh >> 4, h = bh & 15;
    int q0   = qb * 32;
    int lm   = lane & 15, lh = lane >> 4;

    const unsigned short* Qb = Qn + (size_t)bh * SLEN * HDIM;
    const unsigned short* Kb = Kn + (size_t)bh * SLEN * HDIM;
    const unsigned short* Vb = Vt + (size_t)bh * HDIM * SLEN;

    // resident Q fragments: 2 m-tiles x 4 k-chunks
    AFrag qf[2][4];
#pragma unroll
    for (int mi = 0; mi < 2; ++mi)
#pragma unroll
        for (int kc = 0; kc < 4; ++kc)
            load_frag(Qb, (size_t)(q0 + mi * 16), HDIM, kc * 32, lane, qf[mi][kc]);

    v8f o[2][8] = {};
    float mrow[2][8], lrow[2][8];
#pragma unroll
    for (int mi = 0; mi < 2; ++mi)
#pragma unroll
        for (int r = 0; r < 8; ++r) { mrow[mi][r] = -1e30f; lrow[mi][r] = 0.0f; }

    for (int kb0 = 0; kb0 < SLEN; kb0 += 32) {
        // ---- S = Q K^T (32x32) ----
        v8f sc[2][2] = {};
#pragma unroll
        for (int ni = 0; ni < 2; ++ni)
#pragma unroll
            for (int kc = 0; kc < 4; ++kc) {
                AFrag kf;
                load_frag(Kb, (size_t)(kb0 + ni * 16), HDIM, kc * 32, lane, kf);
                sc[0][ni] = wmma_bf16(qf[0][kc].v, kf.v, sc[0][ni]);
                sc[1][ni] = wmma_bf16(qf[1][kc].v, kf.v, sc[1][ni]);
            }

        // ---- online softmax (rows live in 16-lane halves: xor 1,2,4,8) ----
        float pv[2][2][8];
#pragma unroll
        for (int mi = 0; mi < 2; ++mi)
#pragma unroll
            for (int r = 0; r < 8; ++r) {
                float v = fmaxf(sc[mi][0][r], sc[mi][1][r]);
                v = fmaxf(v, __shfl_xor(v, 1));
                v = fmaxf(v, __shfl_xor(v, 2));
                v = fmaxf(v, __shfl_xor(v, 4));
                v = fmaxf(v, __shfl_xor(v, 8));
                float mold = mrow[mi][r];
                float mnew = fmaxf(mold, v);
                float alpha = __expf(mold - mnew);
                mrow[mi][r] = mnew;
                float p0 = __expf(sc[mi][0][r] - mnew);
                float p1 = __expf(sc[mi][1][r] - mnew);
                pv[mi][0][r] = p0; pv[mi][1][r] = p1;
                float rs = p0 + p1;
                rs += __shfl_xor(rs, 1);
                rs += __shfl_xor(rs, 2);
                rs += __shfl_xor(rs, 4);
                rs += __shfl_xor(rs, 8);
                lrow[mi][r] = lrow[mi][r] * alpha + rs;
#pragma unroll
                for (int nh = 0; nh < 8; ++nh) o[mi][nh][r] *= alpha;
            }

        // ---- stage P into LDS (bf16), reload as A fragments ----
#pragma unroll
        for (int mi = 0; mi < 2; ++mi)
#pragma unroll
            for (int ni = 0; ni < 2; ++ni)
#pragma unroll
                for (int r = 0; r < 8; ++r)
                    Pl[(mi * 16 + r + 8 * lh) * 32 + ni * 16 + lm] =
                        f2bf(pv[mi][ni][r]);
        asm volatile("s_wait_dscnt 0x0" ::: "memory");

        AFrag pf[2];
#pragma unroll
        for (int mi = 0; mi < 2; ++mi) {
            const uint4* p4 = reinterpret_cast<const uint4*>(
                Pl + (mi * 16 + lm) * 32 + lh * 8);
            pf[mi].q[0] = p4[0];
            pf[mi].q[1] = p4[2];
        }

        // ---- O += P V (V pre-transposed: contiguous along keys) ----
#pragma unroll
        for (int nh = 0; nh < 8; ++nh) {
            AFrag vf;
            load_frag(Vb, (size_t)(nh * 16), SLEN, kb0, lane, vf);
            o[0][nh] = wmma_bf16(pf[0].v, vf.v, o[0][nh]);
            o[1][nh] = wmma_bf16(pf[1].v, vf.v, o[1][nh]);
        }
        asm volatile("s_wait_dscnt 0x0" ::: "memory");
    }

    // ---- epilogue: O /= l, write bf16 [b, s, h*HDIM + hd] ----
    unsigned short* Ob = Obf + ((size_t)b * SLEN + q0) * DIM + h * HDIM;
#pragma unroll
    for (int mi = 0; mi < 2; ++mi)
#pragma unroll
        for (int r = 0; r < 8; ++r) {
            float inv = 1.0f / lrow[mi][r];
            size_t rowoff = (size_t)(mi * 16 + r + 8 * lh) * DIM;
#pragma unroll
            for (int nh = 0; nh < 8; ++nh)
                Ob[rowoff + nh * 16 + lm] = f2bf(o[mi][nh][r] * inv);
        }
}

// ---------------------------------------------------------------------------
// Host-side launch (stream-ordered; workspace layout = 192 MB total)
// ---------------------------------------------------------------------------
extern "C" void kernel_launch(void* const* d_in, const int* in_sizes, int n_in,
                              void* d_out, int out_size, void* d_ws, size_t ws_size,
                              hipStream_t stream) {
    const float* x     = (const float*)d_in[0];   // [2,2048,2048]
    const float* cosb  = (const float*)d_in[1];   // [2048,128]
    const float* sinb  = (const float*)d_in[2];   // [2048,128]
    const float* Wqkv  = (const float*)d_in[3];   // [6144,2048]
    const float* Wproj = (const float*)d_in[4];   // [2048,2048]
    float* out = (float*)d_out;                   // [2,2048,2048] f32

    char* ws = (char*)d_ws;
    const size_t SZ_XBF   = (size_t)ROWS * DIM * 2;       // 16 MB (reused as attn out)
    const size_t SZ_WQKV  = (size_t)QKVN * DIM * 2;       // 24 MB
    const size_t SZ_WPROJ = (size_t)DIM * DIM * 2;        //  8 MB
    const size_t SZ_QKV   = (size_t)ROWS * QKVN * 4;      // 96 MB
    const size_t SZ_HEAD  = (size_t)BATCH * NHEADS * SLEN * HDIM * 2;  // 16 MB

    unsigned short* xbf   = (unsigned short*)(ws);
    unsigned short* wqkvb = (unsigned short*)(ws + SZ_XBF);
    unsigned short* wprjb = (unsigned short*)(ws + SZ_XBF + SZ_WQKV);
    float*          qkv   = (float*)(ws + SZ_XBF + SZ_WQKV + SZ_WPROJ);
    unsigned short* Qn    = (unsigned short*)(ws + SZ_XBF + SZ_WQKV + SZ_WPROJ + SZ_QKV);
    unsigned short* Kn    = Qn + (SZ_HEAD / 2);
    unsigned short* Vt    = Kn + (SZ_HEAD / 2);
    unsigned short* attnb = xbf;   // alias: x_bf16 is dead after GEMM 1

    // 1) convert to bf16
    {
        int n4 = ROWS * DIM / 4;
        cvt_bf16_kernel<<<(n4 + 255) / 256, 256, 0, stream>>>(x, xbf, n4);
        n4 = QKVN * DIM / 4;
        cvt_bf16_kernel<<<(n4 + 255) / 256, 256, 0, stream>>>(Wqkv, wqkvb, n4);
        n4 = DIM * DIM / 4;
        cvt_bf16_kernel<<<(n4 + 255) / 256, 256, 0, stream>>>(Wproj, wprjb, n4);
    }
    // 2) QKV GEMM: [4096,6144] = x_bf @ Wqkv_bf^T
    {
        dim3 grid(QKVN / 128, ROWS / 128);
        gemm_bf16_nt_kernel<<<grid, 128, 0, stream>>>(xbf, wqkvb, qkv,
                                                      ROWS, QKVN, DIM);
    }
    // 3) RMSNorm + RoPE + V transpose
    rmsrope_kernel<<<BATCH * NHEADS * SLEN, 32, 0, stream>>>(qkv, cosb, sinb,
                                                             Qn, Kn, Vt);
    // 4) flash attention (one wave per 32 query rows per head)
    attn_kernel<<<BATCH * NHEADS * (SLEN / 32), 32, 0, stream>>>(Qn, Kn, Vt, attnb);
    // 5) output projection: [4096,2048] f32 = attn_bf @ Wproj_bf^T
    {
        dim3 grid(DIM / 128, ROWS / 128);
        gemm_bf16_nt_kernel<<<grid, 128, 0, stream>>>(attnb, wprjb, out,
                                                      ROWS, DIM, DIM);
    }
}
